// CategorySpecificInitNet_11218454577237
// MI455X (gfx1250) — matrix operand order repaired and checked by reading, not compiled, not run on hardware
//
#include <hip/hip_runtime.h>

// ---------------------------------------------------------------------------
// Types for CDNA5 WMMA (wave32): v16bf A/B operands, v8f f32 accumulator.
// ---------------------------------------------------------------------------
typedef __attribute__((ext_vector_type(16))) __bf16 v16bf;
typedef __attribute__((ext_vector_type(8)))  float  v8f;

union V16B {
    v16bf v;
    uint4 q[2];
    unsigned short u[16];
};

__device__ __forceinline__ unsigned short f2bf(float f) {
    unsigned u = __builtin_bit_cast(unsigned, f);
    u += 0x7FFFu + ((u >> 16) & 1u);   // round-to-nearest-even
    return (unsigned short)(u >> 16);
}

__device__ __forceinline__ v8f wmma_bf16(v16bf a, v16bf b, v8f c) {
    return __builtin_amdgcn_wmma_f32_16x16x32_bf16(
        false, a, false, b, (short)0, c, false, false);
}

// ---------------------------------------------------------------------------
// CDNA5 async copy: 16 bytes/lane global -> LDS, tracked with ASYNCcnt.
//   global addr = SGPR64 base + VGPR32 offset  (GVS mode)
//   lds addr    = per-lane VGPR byte address
// ---------------------------------------------------------------------------
__device__ __forceinline__ void async_copy16(unsigned lds_addr, unsigned goff,
                                             unsigned long long sbase) {
    asm volatile("global_load_async_to_lds_b128 %0, %1, %2"
                 :: "v"(lds_addr), "v"(goff), "s"(sbase) : "memory");
}
__device__ __forceinline__ void async_wait0() {
    asm volatile("s_wait_asynccnt 0x0" ::: "memory");
}

// A operand (16x32, bf16, row-major source, row stride = 32 elements):
// lane<16 : row=lane,     K = [0..7]  and [16..23]
// lane>=16: row=lane-16,  K = [8..15] and [24..31]
__device__ __forceinline__ v16bf load_a_bf16(const unsigned short* Arow, int kbase, int half) {
    const int c = kbase + half * 8;
    V16B t;
    t.q[0] = *reinterpret_cast<const uint4*>(Arow + c);
    t.q[1] = *reinterpret_cast<const uint4*>(Arow + c + 16);
    return t.v;
}

__device__ __forceinline__ v16bf load_a_f32(const float* Arow, int kbase, int half) {
    const int c = kbase + half * 8;
    V16B t;
#pragma unroll
    for (int j = 0; j < 8; ++j) {
        t.u[j]     = f2bf(Arow[c + j]);
        t.u[8 + j] = f2bf(Arow[c + 16 + j]);
    }
    return t.v;
}

// B operand from pre-packed weights: 32 contiguous bytes per lane.
__device__ __forceinline__ v16bf load_b(const unsigned short* Bp, int nt, int kt, int KT, int lane) {
    const unsigned short* p = Bp + (((size_t)nt * KT + kt) * 32 + lane) * 16;
    V16B t;
    t.q[0] = *reinterpret_cast<const uint4*>(p);
    t.q[1] = *reinterpret_cast<const uint4*>(p + 8);
    return t.v;
}

// ---------------------------------------------------------------------------
// Weight packing: W[K,N] f32 (row-major) -> bf16 in WMMA B-operand tile order.
// ---------------------------------------------------------------------------
__global__ void pack_w(const float* __restrict__ W, unsigned short* __restrict__ out,
                       int Kdim, int Ndim, int batch) {
    const long long idx = (long long)blockIdx.x * blockDim.x + threadIdx.x;
    const long long per = (long long)Kdim * Ndim;
    if (idx >= per * batch) return;
    const int b = (int)(idx / per);
    const int r = (int)(idx - (long long)b * per);
    const int j    = r & 15;
    const int lane = (r >> 4) & 31;
    const int tile = r >> 9;
    const int KT = Kdim >> 5;
    const int kt = tile % KT;
    const int nt = tile / KT;
    const int col  = nt * 16 + (lane & 15);
    const int krow = kt * 32 + (lane >> 4) * 8 + (j & 7) + (j >> 3) * 16;
    out[idx] = f2bf(W[(size_t)b * per + (size_t)krow * Ndim + col]);
}

// ---------------------------------------------------------------------------
// Bucketing kernels (route rows by category).
// ---------------------------------------------------------------------------
__global__ void zero16(int* p) {
    if (threadIdx.x < 16) p[threadIdx.x] = 0;
}
__global__ void hist_k(const int* __restrict__ cat, int* __restrict__ counts) {
    const int i = blockIdx.x * 256 + threadIdx.x;
    atomicAdd(&counts[cat[i] & 7], 1);
}
__global__ void prefix8(const int* __restrict__ counts, int* __restrict__ offsets) {
    if (threadIdx.x == 0) {
        int s = 0;
#pragma unroll
        for (int i = 0; i < 8; ++i) { offsets[i] = s; s += counts[i]; }
    }
}
__global__ void scatter_k(const int* __restrict__ cat, const int* __restrict__ offsets,
                          int* __restrict__ cursors, int* __restrict__ brows) {
    const int i = blockIdx.x * 256 + threadIdx.x;
    const int k = cat[i] & 7;
    const int pos = atomicAdd(&cursors[k], 1);
    brows[offsets[k] + pos] = i;
}

// ---------------------------------------------------------------------------
// Encoder GEMM: Out[B,N](bf16) = act(A[B,K] @ W[K,N] + bias)
// 256 threads = 8 waves, all sharing one 16-row A tile (async-staged in LDS,
// double-buffered). Each wave owns a 16(M) x 32(N) output tile.
// ---------------------------------------------------------------------------
template <bool AF32, bool RELU>
__global__ void enc_gemm(const void* __restrict__ Ain, const unsigned short* __restrict__ Bp,
                         const float* __restrict__ bias, unsigned short* __restrict__ Out,
                         int Kdim, int Ndim, int groups) {
    constexpr int ES   = AF32 ? 4 : 2;      // element size of A
    constexpr int BUFB = 16 * 32 * ES;      // bytes per stage buffer
    constexpr int NI   = BUFB / 512;        // async instrs per stage (32 lanes x 16B)
    constexpr int CPR  = (32 * ES) / 16;    // 16B chunks per A row
    __shared__ __align__(16) unsigned char stage[2][BUFB];

    const int lane = threadIdx.x & 31;
    const int wave = threadIdx.x >> 5;
    const int mt = blockIdx.x / groups;
    const int g  = blockIdx.x - mt * groups;
    const int n2 = g * 8 + wave;
    const int ln = lane & 15;
    const int half = lane >> 4;
    const int KT = Kdim >> 5;
    const int nt0 = n2 * 2, nt1 = nt0 + 1;

    const unsigned long long abase =
        (unsigned long long)(uintptr_t)Ain + (unsigned long long)(mt * 16) * Kdim * ES;
    const int pitch = Kdim * ES;

    auto issue = [&](int kt, int buf) {
        if (wave == 0) {
#pragma unroll
            for (int i = 0; i < NI; ++i) {
                const int chunk  = i * 32 + lane;
                const int row    = chunk / CPR;
                const int within = chunk - row * CPR;
                const unsigned lds  = (unsigned)(uintptr_t)&stage[buf][chunk * 16];
                const unsigned goff = (unsigned)(row * pitch + kt * 32 * ES + within * 16);
                async_copy16(lds, goff, abase);
            }
        }
    };

    issue(0, 0);
    v8f acc0 = {}, acc1 = {};
    for (int kt = 0; kt < KT; ++kt) {
        async_wait0();            // non-issuing waves: ASYNCcnt already 0
        __syncthreads();          // staged slice kt visible to all waves
        if (kt + 1 < KT) issue(kt + 1, (kt + 1) & 1);
        const unsigned char* buf = stage[kt & 1];
        v16bf a;
        if (AF32) a = load_a_f32((const float*)buf + ln * 32, 0, half);
        else      a = load_a_bf16((const unsigned short*)buf + ln * 32, 0, half);
        const v16bf b0 = load_b(Bp, nt0, kt, KT, lane);
        const v16bf b1 = load_b(Bp, nt1, kt, KT, lane);
        acc0 = wmma_bf16(a, b0, acc0);
        acc1 = wmma_bf16(a, b1, acc1);
    }
    const float bb0 = bias[nt0 * 16 + ln];
    const float bb1 = bias[nt1 * 16 + ln];
#pragma unroll
    for (int r = 0; r < 8; ++r) {
        const int orow = mt * 16 + half * 8 + r;
        float v0 = acc0[r] + bb0;
        float v1 = acc1[r] + bb1;
        if (RELU) { v0 = fmaxf(v0, 0.f); v1 = fmaxf(v1, 0.f); }
        Out[(size_t)orow * Ndim + nt0 * 16 + ln] = f2bf(v0);
        Out[(size_t)orow * Ndim + nt1 * 16 + ln] = f2bf(v1);
    }
}

// ---------------------------------------------------------------------------
// Fused routed decoder: one block = 16 rows of one category bucket.
// Gathered input rows async-staged into LDS once; 3 layers of WMMA with
// activations kept in LDS (bf16).
// ---------------------------------------------------------------------------
__global__ void dec_fused(const unsigned short* __restrict__ H3,
                          const unsigned short* __restrict__ Wd1p, const float* __restrict__ bd1,
                          const unsigned short* __restrict__ Wd2p, const float* __restrict__ bd2,
                          const unsigned short* __restrict__ Wd3p, const float* __restrict__ bd3,
                          const int* __restrict__ counts, const int* __restrict__ offsets,
                          const int* __restrict__ brows, float* __restrict__ out) {
    __shared__ __align__(16) unsigned short act0[16 * 256];  // gathered H3 rows
    __shared__ __align__(16) unsigned short act1[16 * 256];
    __shared__ __align__(16) unsigned short act2[16 * 256];
    __shared__ int rows_s[16];

    const int k  = blockIdx.x >> 11;
    const int tt = blockIdx.x & 2047;
    const int cnt = counts[k];
    if (tt * 16 >= cnt) return;           // block-uniform exit
    const int off = offsets[k];
    if (threadIdx.x < 16) {
        const int i = tt * 16 + threadIdx.x;
        rows_s[threadIdx.x] = brows[off + (i < cnt ? i : 0)];
    }
    __syncthreads();

    const int lane = threadIdx.x & 31;
    const int wave = threadIdx.x >> 5;
    const int ln = lane & 15;
    const int half = lane >> 4;
    const int nt0 = wave * 2, nt1 = nt0 + 1;

    // Async row gather: 16 rows x 512B; instruction i copies row i (32 lanes x 16B).
    if (wave == 0) {
        const unsigned long long hbase = (unsigned long long)(uintptr_t)H3;
#pragma unroll
        for (int i = 0; i < 16; ++i) {
            const unsigned lds  = (unsigned)(uintptr_t)&act0[i * 256] + lane * 16;
            const unsigned goff = (unsigned)rows_s[i] * 512u + lane * 16;
            async_copy16(lds, goff, hbase);
        }
    }
    async_wait0();
    __syncthreads();

    // ---- layer 1: act0 @ Wd1[k] + bd1[k], relu -> act1 ----
    {
        const unsigned short* Arow = &act0[ln * 256];
        const unsigned short* W = Wd1p + (size_t)k * 65536;
        v8f a0 = {}, a1 = {};
#pragma unroll
        for (int kt = 0; kt < 8; ++kt) {
            const v16bf a  = load_a_bf16(Arow, kt * 32, half);
            const v16bf b0 = load_b(W, nt0, kt, 8, lane);
            const v16bf b1 = load_b(W, nt1, kt, 8, lane);
            a0 = wmma_bf16(a, b0, a0);
            a1 = wmma_bf16(a, b1, a1);
        }
        const float bb0 = bd1[k * 256 + nt0 * 16 + ln];
        const float bb1 = bd1[k * 256 + nt1 * 16 + ln];
#pragma unroll
        for (int r = 0; r < 8; ++r) {
            const int rr = half * 8 + r;
            act1[rr * 256 + nt0 * 16 + ln] = f2bf(fmaxf(a0[r] + bb0, 0.f));
            act1[rr * 256 + nt1 * 16 + ln] = f2bf(fmaxf(a1[r] + bb1, 0.f));
        }
    }
    __syncthreads();

    // ---- layer 2: act1 @ Wd2[k] + bd2[k], relu -> act2 ----
    {
        const unsigned short* Arow = &act1[ln * 256];
        const unsigned short* W = Wd2p + (size_t)k * 65536;
        v8f a0 = {}, a1 = {};
#pragma unroll
        for (int kt = 0; kt < 8; ++kt) {
            const v16bf a  = load_a_bf16(Arow, kt * 32, half);
            const v16bf b0 = load_b(W, nt0, kt, 8, lane);
            const v16bf b1 = load_b(W, nt1, kt, 8, lane);
            a0 = wmma_bf16(a, b0, a0);
            a1 = wmma_bf16(a, b1, a1);
        }
        const float bb0 = bd2[k * 256 + nt0 * 16 + ln];
        const float bb1 = bd2[k * 256 + nt1 * 16 + ln];
#pragma unroll
        for (int r = 0; r < 8; ++r) {
            const int rr = half * 8 + r;
            act2[rr * 256 + nt0 * 16 + ln] = f2bf(fmaxf(a0[r] + bb0, 0.f));
            act2[rr * 256 + nt1 * 16 + ln] = f2bf(fmaxf(a1[r] + bb1, 0.f));
        }
    }
    __syncthreads();

    // ---- layer 3: act2 @ Wd3[k] + bd3[k] -> d_out (f32 scatter) ----
    {
        const unsigned short* Arow = &act2[ln * 256];
        const unsigned short* W = Wd3p + (size_t)k * 32768;
        v8f a0 = {};
#pragma unroll
        for (int kt = 0; kt < 8; ++kt) {
            const v16bf a = load_a_bf16(Arow, kt * 32, half);
            const v16bf b = load_b(W, wave, kt, 8, lane);
            a0 = wmma_bf16(a, b, a0);
        }
        const float bb = bd3[k * 128 + wave * 16 + ln];
#pragma unroll
        for (int r = 0; r < 8; ++r) {
            const int rr = half * 8 + r;
            if (tt * 16 + rr < cnt)
                out[(size_t)rows_s[rr] * 128 + wave * 16 + ln] = a0[r] + bb;
        }
    }
}

// ---------------------------------------------------------------------------
// Host launcher
// ---------------------------------------------------------------------------
extern "C" void kernel_launch(void* const* d_in, const int* in_sizes, int n_in,
                              void* d_out, int out_size, void* d_ws, size_t ws_size,
                              hipStream_t stream) {
    (void)in_sizes; (void)n_in; (void)out_size; (void)ws_size;
    const float* features = (const float*)d_in[0];
    const float* We1 = (const float*)d_in[1];  const float* be1 = (const float*)d_in[2];
    const float* We2 = (const float*)d_in[3];  const float* be2 = (const float*)d_in[4];
    const float* We3 = (const float*)d_in[5];  const float* be3 = (const float*)d_in[6];
    const float* Wd1 = (const float*)d_in[7];  const float* bd1 = (const float*)d_in[8];
    const float* Wd2 = (const float*)d_in[9];  const float* bd2 = (const float*)d_in[10];
    const float* Wd3 = (const float*)d_in[11]; const float* bd3 = (const float*)d_in[12];
    const int*   cat = (const int*)d_in[13];
    float* out = (float*)d_out;

    // Workspace layout
    char* ws = (char*)d_ws;
    int* counts  = (int*)ws;          // [8]
    int* cursors = counts + 8;        // [8]
    int* offsets = counts + 16;       // [8]
    int* brows   = counts + 32;       // [32768]
    unsigned short* h = (unsigned short*)(ws + 256 * 1024);
    size_t o = 0;
    unsigned short* We1p = h + o; o += (size_t)768 * 512;
    unsigned short* We2p = h + o; o += (size_t)512 * 256;
    unsigned short* We3p = h + o; o += (size_t)256 * 256;
    unsigned short* Wd1p = h + o; o += (size_t)8 * 256 * 256;
    unsigned short* Wd2p = h + o; o += (size_t)8 * 256 * 256;
    unsigned short* Wd3p = h + o; o += (size_t)8 * 256 * 128;
    unsigned short* H1   = h + o; o += (size_t)32768 * 512;
    unsigned short* H2   = h + o; o += (size_t)32768 * 256;
    unsigned short* H3   = h + o; o += (size_t)32768 * 256;

    // Routing
    zero16<<<1, 32, 0, stream>>>(counts);
    hist_k<<<32768 / 256, 256, 0, stream>>>(cat, counts);
    prefix8<<<1, 32, 0, stream>>>(counts, offsets);
    scatter_k<<<32768 / 256, 256, 0, stream>>>(cat, offsets, cursors, brows);

    // Weight packing (f32 -> bf16 B-operand tiles)
    pack_w<<<(768 * 512 + 255) / 256, 256, 0, stream>>>(We1, We1p, 768, 512, 1);
    pack_w<<<(512 * 256 + 255) / 256, 256, 0, stream>>>(We2, We2p, 512, 256, 1);
    pack_w<<<(256 * 256 + 255) / 256, 256, 0, stream>>>(We3, We3p, 256, 256, 1);
    pack_w<<<(8 * 256 * 256 + 255) / 256, 256, 0, stream>>>(Wd1, Wd1p, 256, 256, 8);
    pack_w<<<(8 * 256 * 256 + 255) / 256, 256, 0, stream>>>(Wd2, Wd2p, 256, 256, 8);
    pack_w<<<(8 * 256 * 128 + 255) / 256, 256, 0, stream>>>(Wd3, Wd3p, 256, 128, 8);

    // Encoder: 32768 rows, blocks share one A tile across 8 waves.
    // E1: 2048 M-tiles x (512/32/8 = 2 groups) = 4096 blocks
    enc_gemm<true,  true ><<<4096, 256, 0, stream>>>(features, We1p, be1, H1, 768, 512, 2);
    // E2/E3: 2048 M-tiles x 1 group
    enc_gemm<false, true ><<<2048, 256, 0, stream>>>(H1, We2p, be2, H2, 512, 256, 1);
    enc_gemm<false, false><<<2048, 256, 0, stream>>>(H2, We3p, be3, H3, 256, 256, 1);

    // Routed fused decoder: 8 categories x 2048 worst-case tiles
    dec_fused<<<8 * 2048, 256, 0, stream>>>(H3, Wd1p, bd1, Wd2p, bd2, Wd3p, bd3,
                                            counts, offsets, brows, out);
}
